// TransformerBlock_88321707475317
// MI455X (gfx1250) — matrix operand order repaired and checked by reading, not compiled, not run on hardware
//
#include <hip/hip_runtime.h>
#include <hip/hip_bf16.h>
#include <math.h>

// ---- problem constants (from reference) ----
#define E_DIM 768
#define H_NUM 12
#define D_DIM 64
#define F_DIM 3072
#define SEQ   2048
#define BATCH 8
#define M_TOK (BATCH*SEQ)   // 16384 tokens

typedef __attribute__((ext_vector_type(16))) __bf16 v16bf;
typedef __attribute__((ext_vector_type(8)))  float  v8f;

union FragAB {
    v16bf v;
    uint4 q[2];
};

// ---- CDNA5 async global->LDS staging (ASYNCcnt-tracked), guarded ----
#if __has_builtin(__builtin_amdgcn_global_load_async_to_lds_b128)
#define ASYNC_LDS 1
typedef int v4i __attribute__((ext_vector_type(4)));
typedef __attribute__((address_space(1))) v4i as1_v4i;   // global
typedef __attribute__((address_space(3))) v4i as3_v4i;   // LDS
__device__ __forceinline__ void async_ld_b128(const void* g, void* l) {
    __builtin_amdgcn_global_load_async_to_lds_b128((as1_v4i*)g, (as3_v4i*)l, 0, 0);
}
__device__ __forceinline__ void async_wait0() {
#if __has_builtin(__builtin_amdgcn_s_wait_asynccnt)
    __builtin_amdgcn_s_wait_asynccnt(0);
#else
    asm volatile("s_wait_asynccnt 0" ::: "memory");
#endif
}
#else
__device__ __forceinline__ void async_wait0() {}
#endif

// ---------------- fp32 -> bf16 cast (vectorized) ----------------
__global__ __launch_bounds__(256) void cast_f32_to_bf16(
    const float* __restrict__ s, __bf16* __restrict__ d, int n)
{
    int i = (blockIdx.x * blockDim.x + threadIdx.x) * 4;
    if (i + 3 < n) {
        float4 f = *(const float4*)(s + i);
        d[i + 0] = (__bf16)f.x;
        d[i + 1] = (__bf16)f.y;
        d[i + 2] = (__bf16)f.z;
        d[i + 3] = (__bf16)f.w;
    }
}

// ---------------- WMMA GEMM: C[M x N] = A[M x K] * W[N x K]^T + bias ----------------
// A, W bf16 row-major.  EPI: 0 = store bf16, 1 = store fp32, 2 = exact GELU -> bf16.
// Block: 256 threads (8 waves).  Tile BM=BN=128, KT=32.  Wave tile 32x64 (2x4 WMMA).
template <int EPI>
__global__ __launch_bounds__(256) void gemm_bf16_wmma(
    const __bf16* __restrict__ A, const __bf16* __restrict__ W,
    const float* __restrict__ bias, void* __restrict__ out,
    int K, int ldc)
{
    constexpr int BM = 128, BN = 128, KT = 32;
    constexpr int LS = 40;  // LDS row pitch in bf16: 80B -> 16B aligned, conflict-friendly
    __shared__ __align__(16) __bf16 As[2][BM * LS];
    __shared__ __align__(16) __bf16 Bs[2][BN * LS];

    const int tid  = threadIdx.x;
    const int lane = tid & 31;
    const int wave = tid >> 5;
    const int half = lane >> 4;   // lane 0-15 vs 16-31
    const int l16  = lane & 15;
    const int wm   = wave & 3;    // 4 waves along M
    const int wn   = wave >> 2;   // 2 waves along N
    const long tileM = (long)blockIdx.x * BM;
    const long tileN = (long)blockIdx.y * BN;

    v8f c[2][4];
#pragma unroll
    for (int i = 0; i < 2; ++i)
#pragma unroll
        for (int j = 0; j < 4; ++j) c[i][j] = {};

    auto stage = [&](int buf, int k0) {
        // 128 rows x 32 bf16 = 512 x 16B per operand; 2 chunks per thread each
        for (int t = tid; t < BM * 4; t += 256) {
            int r = t >> 2, ch = t & 3;
#ifdef ASYNC_LDS
            async_ld_b128(&A[(tileM + r) * (long)K + k0 + ch * 8],
                          &As[buf][r * LS + ch * 8]);
#else
            *(uint4*)&As[buf][r * LS + ch * 8] =
                *(const uint4*)&A[(tileM + r) * (long)K + k0 + ch * 8];
#endif
        }
        for (int t = tid; t < BN * 4; t += 256) {
            int r = t >> 2, ch = t & 3;
#ifdef ASYNC_LDS
            async_ld_b128(&W[(tileN + r) * (long)K + k0 + ch * 8],
                          &Bs[buf][r * LS + ch * 8]);
#else
            *(uint4*)&Bs[buf][r * LS + ch * 8] =
                *(const uint4*)&W[(tileN + r) * (long)K + k0 + ch * 8];
#endif
        }
    };

    const int nk = K / KT;
    stage(0, 0);
    int cur = 0;
    for (int kb = 0; kb < nk; ++kb) {
        async_wait0();       // own async LDS writes visible before barrier
        __syncthreads();
        if (kb + 1 < nk) {
            if (kb + 2 < nk) {
                // L2 prefetch of tile-after-next (global_prefetch_b8)
                __builtin_prefetch(&A[(tileM + (tid >> 1)) * (long)K + (kb + 2) * KT + (tid & 1) * 16], 0, 1);
                __builtin_prefetch(&W[(tileN + (tid >> 1)) * (long)K + (kb + 2) * KT + (tid & 1) * 16], 0, 1);
            }
            stage(cur ^ 1, (kb + 1) * KT);
        }
        // A fragment (16x32 bf16): lane = row, VGPR0-3: K = half*8 + {0..7},
        // VGPR4-7: K = 16 + half*8 + {0..7}  -> two contiguous b128 LDS reads
        FragAB a[2], b[4];
#pragma unroll
        for (int tm = 0; tm < 2; ++tm) {
            const __bf16* ap = &As[cur][(wm * 32 + tm * 16 + l16) * LS];
            a[tm].q[0] = *(const uint4*)(ap + half * 8);
            a[tm].q[1] = *(const uint4*)(ap + 16 + half * 8);
        }
        // B fragment (32x16 bf16): lane = col, same K order as A -> two b128 LDS reads
#pragma unroll
        for (int tn = 0; tn < 4; ++tn) {
            const __bf16* bp = &Bs[cur][(wn * 64 + tn * 16 + l16) * LS];
            b[tn].q[0] = *(const uint4*)(bp + half * 8);
            b[tn].q[1] = *(const uint4*)(bp + 16 + half * 8);
        }
#pragma unroll
        for (int tm = 0; tm < 2; ++tm)
#pragma unroll
            for (int tn = 0; tn < 4; ++tn)
                c[tm][tn] = __builtin_amdgcn_wmma_f32_16x16x32_bf16(
                    false, a[tm].v, false, b[tn].v, (short)0, c[tm][tn], false, false);
        cur ^= 1;
    }

    // Epilogue: C 16x16 f32 layout -> lane col = l16, VGPR v -> row v + half*8
#pragma unroll
    for (int tn = 0; tn < 4; ++tn) {
        const long gn = tileN + wn * 64 + tn * 16 + l16;
        const float bv = bias[gn];
#pragma unroll
        for (int tm = 0; tm < 2; ++tm) {
#pragma unroll
            for (int v = 0; v < 8; ++v) {
                const long gm = tileM + wm * 32 + tm * 16 + half * 8 + v;
                float val = c[tm][tn][v] + bv;
                if (EPI == 2)
                    val = 0.5f * val * (1.f + erff(val * 0.70710678118654752f));
                if (EPI == 1)
                    ((float*)out)[gm * ldc + gn] = val;
                else
                    ((__bf16*)out)[gm * ldc + gn] = (__bf16)val;
            }
        }
    }
}

// ---------------- per-token head-softmax attention ----------------
// energy[i][j] = <q_i, k_j> / sqrt(E)  (12x12), softmax over j,
// ctx[n, i, s, d] = sum_j p[i][j] * v[j][d]   (written in (N,H,S,D) flat order,
// matching the reference's reshape of the 'nqhd' einsum output)
__global__ __launch_bounds__(128) void attn_per_token(
    const __bf16* __restrict__ Q, const __bf16* __restrict__ K,
    const __bf16* __restrict__ V, __bf16* __restrict__ ctx)
{
    __shared__ float qs[E_DIM], ks[E_DIM], vs[E_DIM];
    __shared__ float es[H_NUM * H_NUM];
    const int  m    = blockIdx.x;
    const int  tid  = threadIdx.x;
    const long base = (long)m * E_DIM;

    for (int i = tid; i < E_DIM; i += 128) {
        qs[i] = (float)Q[base + i];
        ks[i] = (float)K[base + i];
        vs[i] = (float)V[base + i];
    }
    __syncthreads();

    const float scale = 0.03608439182435161f;  // 1/sqrt(768)
    for (int e = tid; e < H_NUM * H_NUM; e += 128) {
        int i = e / H_NUM, j = e % H_NUM;
        float acc = 0.f;
#pragma unroll 8
        for (int d = 0; d < D_DIM; ++d) acc += qs[i * D_DIM + d] * ks[j * D_DIM + d];
        es[e] = acc * scale;
    }
    __syncthreads();

    if (tid < H_NUM) {
        float mx = -1e30f;
        for (int j = 0; j < H_NUM; ++j) mx = fmaxf(mx, es[tid * H_NUM + j]);
        float p[H_NUM], sum = 0.f;
        for (int j = 0; j < H_NUM; ++j) { p[j] = __expf(es[tid * H_NUM + j] - mx); sum += p[j]; }
        float inv = 1.f / sum;
        for (int j = 0; j < H_NUM; ++j) es[tid * H_NUM + j] = p[j] * inv;
    }
    __syncthreads();

    const int n = m / SEQ, s = m % SEQ;
    for (int o = tid; o < E_DIM; o += 128) {
        int i = o / D_DIM, d = o % D_DIM;
        float acc = 0.f;
#pragma unroll
        for (int j = 0; j < H_NUM; ++j) acc += es[i * H_NUM + j] * vs[j * D_DIM + d];
        ctx[(((long)n * H_NUM + i) * SEQ + s) * D_DIM + d] = (__bf16)acc;
    }
}

// ---------------- fused residual add + LayerNorm ----------------
// out = LN(a + b) * gamma + beta ; fp32 always, bf16 copy if outb != nullptr
__global__ __launch_bounds__(256) void add_layernorm(
    const float* __restrict__ a, const float* __restrict__ b,
    const float* __restrict__ gamma, const float* __restrict__ beta,
    float* __restrict__ outf, __bf16* __restrict__ outb)
{
    __shared__ float red[256];
    const int  tid  = threadIdx.x;
    const long base = (long)blockIdx.x * E_DIM;

    float x0 = a[base + tid]       + b[base + tid];
    float x1 = a[base + tid + 256] + b[base + tid + 256];
    float x2 = a[base + tid + 512] + b[base + tid + 512];

    red[tid] = x0 + x1 + x2;
    __syncthreads();
    for (int st = 128; st > 0; st >>= 1) { if (tid < st) red[tid] += red[tid + st]; __syncthreads(); }
    const float mu = red[0] * (1.f / E_DIM);
    __syncthreads();

    const float d0 = x0 - mu, d1 = x1 - mu, d2 = x2 - mu;
    red[tid] = d0 * d0 + d1 * d1 + d2 * d2;
    __syncthreads();
    for (int st = 128; st > 0; st >>= 1) { if (tid < st) red[tid] += red[tid + st]; __syncthreads(); }
    const float rstd = rsqrtf(red[0] * (1.f / E_DIM) + 1e-5f);

    float y0 = d0 * rstd * gamma[tid]       + beta[tid];
    float y1 = d1 * rstd * gamma[tid + 256] + beta[tid + 256];
    float y2 = d2 * rstd * gamma[tid + 512] + beta[tid + 512];
    outf[base + tid]       = y0;
    outf[base + tid + 256] = y1;
    outf[base + tid + 512] = y2;
    if (outb) {
        outb[base + tid]       = (__bf16)y0;
        outb[base + tid + 256] = (__bf16)y1;
        outb[base + tid + 512] = (__bf16)y2;
    }
}

// ---------------- host-side orchestration ----------------
extern "C" void kernel_launch(void* const* d_in, const int* in_sizes, int n_in,
                              void* d_out, int out_size, void* d_ws, size_t ws_size,
                              hipStream_t stream)
{
    (void)in_sizes; (void)n_in; (void)out_size; (void)ws_size;

    const float* x   = (const float*)d_in[0];
    const float* Wq  = (const float*)d_in[1];
    const float* bq  = (const float*)d_in[2];
    const float* Wk  = (const float*)d_in[3];
    const float* bk  = (const float*)d_in[4];
    const float* Wv  = (const float*)d_in[5];
    const float* bv  = (const float*)d_in[6];
    const float* Wo  = (const float*)d_in[7];
    const float* bo  = (const float*)d_in[8];
    const float* g1  = (const float*)d_in[9];
    const float* be1 = (const float*)d_in[10];
    const float* W1  = (const float*)d_in[11];
    const float* c1  = (const float*)d_in[12];
    const float* W2  = (const float*)d_in[13];
    const float* c2  = (const float*)d_in[14];
    const float* g2  = (const float*)d_in[15];
    const float* be2 = (const float*)d_in[16];

    // workspace arena (peak ~366 MB); fff aliases the dead Qb+Kb region
    char*  ws  = (char*)d_ws;
    size_t off = 0;
    auto alloc = [&](size_t bytes) -> void* {
        void* p = ws + off;
        off = (off + bytes + 255) & ~(size_t)255;
        return p;
    };
    const size_t ME = (size_t)M_TOK * E_DIM;
    const size_t MF = (size_t)M_TOK * F_DIM;

    __bf16* xb    = (__bf16*)alloc(ME * 2);
    __bf16* wqb   = (__bf16*)alloc((size_t)E_DIM * E_DIM * 2);
    __bf16* wkb   = (__bf16*)alloc((size_t)E_DIM * E_DIM * 2);
    __bf16* wvb   = (__bf16*)alloc((size_t)E_DIM * E_DIM * 2);
    __bf16* wob   = (__bf16*)alloc((size_t)E_DIM * E_DIM * 2);
    __bf16* w1b   = (__bf16*)alloc((size_t)F_DIM * E_DIM * 2);
    __bf16* w2b   = (__bf16*)alloc((size_t)E_DIM * F_DIM * 2);
    __bf16* Qb    = (__bf16*)alloc(ME * 2);
    __bf16* Kb    = (__bf16*)alloc(ME * 2);
    __bf16* Vb    = (__bf16*)alloc(ME * 2);
    __bf16* ctxb  = (__bf16*)alloc(ME * 2);
    float*  attnf = (float*) alloc(ME * 4);
    float*  x1f   = (float*) alloc(ME * 4);
    __bf16* x1b   = (__bf16*)alloc(ME * 2);
    __bf16* hb    = (__bf16*)alloc(MF * 2);
    float*  fff   = (float*)Qb;  // Q/K dead after attention; 2*(ME*2) >= ME*4

    // 1) precision casts
    auto cast = [&](const float* s, __bf16* d, size_t n) {
        cast_f32_to_bf16<<<dim3((unsigned)(n / 1024)), dim3(256), 0, stream>>>(s, d, (int)n);
    };
    cast(x,  xb,  ME);
    cast(Wq, wqb, (size_t)E_DIM * E_DIM);
    cast(Wk, wkb, (size_t)E_DIM * E_DIM);
    cast(Wv, wvb, (size_t)E_DIM * E_DIM);
    cast(Wo, wob, (size_t)E_DIM * E_DIM);
    cast(W1, w1b, (size_t)F_DIM * E_DIM);
    cast(W2, w2b, (size_t)E_DIM * F_DIM);

    const dim3 blk(256);
    const dim3 gEE(M_TOK / 128, E_DIM / 128);  // 128 x 6
    const dim3 gEF(M_TOK / 128, F_DIM / 128);  // 128 x 24

    // 2) QKV projections (bf16 out)
    gemm_bf16_wmma<0><<<gEE, blk, 0, stream>>>(xb, wqb, bq, Qb, E_DIM, E_DIM);
    gemm_bf16_wmma<0><<<gEE, blk, 0, stream>>>(xb, wkb, bk, Kb, E_DIM, E_DIM);
    gemm_bf16_wmma<0><<<gEE, blk, 0, stream>>>(xb, wvb, bv, Vb, E_DIM, E_DIM);

    // 3) per-token softmax-over-heads attention, scrambled (N,H,S,D) ctx layout
    attn_per_token<<<dim3(M_TOK), dim3(128), 0, stream>>>(Qb, Kb, Vb, ctxb);

    // 4) output projection (fp32 out)
    gemm_bf16_wmma<1><<<gEE, blk, 0, stream>>>(ctxb, wob, bo, attnf, E_DIM, E_DIM);

    // 5) x1 = LN(x + attn_out)
    add_layernorm<<<dim3(M_TOK), blk, 0, stream>>>(x, attnf, g1, be1, x1f, x1b);

    // 6) FFN: h = GELU(x1 @ W1^T + c1)   (fused GELU, bf16 out)
    gemm_bf16_wmma<2><<<gEF, blk, 0, stream>>>(x1b, w1b, c1, hb, E_DIM, F_DIM);

    // 7) ff = h @ W2^T + c2 (fp32 out)
    gemm_bf16_wmma<1><<<gEE, blk, 0, stream>>>(hb, w2b, c2, fff, F_DIM, E_DIM);

    // 8) out = LN(x1 + ff)
    add_layernorm<<<dim3(M_TOK), blk, 0, stream>>>(x1f, fff, g2, be2, (float*)d_out, nullptr);
}
